// DeepQNet_20194936226136
// MI455X (gfx1250) — matrix operand order repaired and checked by reading, not compiled, not run on hardware
//
#include <hip/hip_runtime.h>
#include <math.h>

// ---------------- problem constants (match reference) ----------------
#define NN   10000      // nodes
#define BB   4          // batch
#define DD   128        // feature dim (H=1)
#define MROWS (BB*NN)   // 40000 GEMM rows

typedef __attribute__((ext_vector_type(16))) _Float16 v16h;
typedef __attribute__((ext_vector_type(8)))  _Float16 v8h;
typedef __attribute__((ext_vector_type(8)))  float    v8f;

// ---------------- monotone float<->uint map for atomic max ----------------
__device__ __forceinline__ unsigned f2mono(float f) {
    unsigned u = __float_as_uint(f);
    return (u & 0x80000000u) ? ~u : (u | 0x80000000u);
}
__device__ __forceinline__ float mono2f(unsigned m) {
    unsigned u = (m & 0x80000000u) ? (m & 0x7fffffffu) : ~m;
    return __uint_as_float(u);
}

// ---------------- clears ----------------
__global__ void clear_f32(float* p, int n) {
    int i = blockIdx.x * blockDim.x + threadIdx.x;
    if (i < n) p[i] = 0.0f;
}
__global__ void clear_u32(unsigned* p, int n) {
    int i = blockIdx.x * blockDim.x + threadIdx.x;
    if (i < n) p[i] = 0u;   // mono lower bound (self-loops guarantee overwrite)
}
__global__ void clear_f32_v4(float4* p, int n4) {   // coalesced b128 clears
    int i = blockIdx.x * blockDim.x + threadIdx.x;
    if (i < n4) p[i] = make_float4(0.f, 0.f, 0.f, 0.f);
}

// ---------------- fused GEMM: h = (x [+ state*Wns + bns]) @ W [+ biasOut] ---
// A is [MROWS, DD] f32 (xBroadcast: x is [NN,DD] shared across batch),
// W is [DD, DD] f32 row-major. Computed via v_wmma_f32_16x16x32_f16.
// Block: 128 threads = 4 waves; each wave does 16 rows x 128 cols.
__global__ void __launch_bounds__(128)
fused_gemm(const float* __restrict__ xin, int xBroadcast,
           const float* __restrict__ state, const float* __restrict__ Wns,
           const float* __restrict__ bns,   const float* __restrict__ Wmat,
           const float* __restrict__ biasOut, float* __restrict__ hOut)
{
    __shared__ __align__(32) _Float16 Wt[DD * DD];     // W transposed, f16: 32 KB
    __shared__ __align__(32) _Float16 As[4][16 * DD];  // per-wave A stripes: 16 KB

    const int tid  = threadIdx.x;
    const int w    = tid >> 5;          // wave id 0..3
    const int lane = tid & 31;
    const int r    = lane & 15;
    const int sel  = (lane >> 4) & 1;   // 0: lanes 0-15, 1: lanes 16-31
    const int rb   = blockIdx.x * 64 + w * 16;  // first row of this wave

    // ---- stage W transposed into LDS as f16: Wt[col*DD + k] = W[k*DD + col]
    #pragma unroll 4
    for (int k = 0; k < DD; ++k)
        Wt[tid * DD + k] = (_Float16)Wmat[k * DD + tid];   // tid == col (128 threads)

    // ---- stage this wave's 16 fused A rows into LDS as f16
    for (int idx = lane; idx < 16 * DD; idx += 32) {
        int rr = idx >> 7, k = idx & (DD - 1);
        int m  = rb + rr;                       // < MROWS by construction
        int b  = m / NN, n = m - b * NN;
        float v = xBroadcast ? xin[(size_t)n * DD + k]
                             : xin[(size_t)m * DD + k];
        if (Wns) v += state[b * NN + n] * Wns[k] + bns[k];
        As[w][idx] = (_Float16)v;
    }
    __syncthreads();

    // ---- WMMA main loop: 4 k-tiles (K=32 each) x 8 col-tiles.
    // Preload all 8 B fragments per k-tile so the 8 WMMAs don't serialize on
    // per-fragment s_wait_dscnt (LDS latency overlapped with matrix issue).
    const _Float16* Aw = As[w];
    v8f acc[8];
    #pragma unroll
    for (int cc = 0; cc < 8; ++cc) acc[cc] = (v8f)(0.0f);

    #pragma unroll
    for (int t = 0; t < 4; ++t) {
        const int k0 = t * 32;
        // A fragment: lanes<16 hold K={k0..k0+15}; lanes>=16 hold K={k0+16..k0+31}
        v8h lo = *(const v8h*)(Aw + r * DD + k0 + sel * 16);
        v8h hi = *(const v8h*)(Aw + r * DD + k0 + 8 + sel * 16);
        v16h afrag = __builtin_shufflevector(lo, hi,
                         0,1,2,3,4,5,6,7,8,9,10,11,12,13,14,15);
        v16h bfr[8];
        #pragma unroll
        for (int cc = 0; cc < 8; ++cc)
            bfr[cc] = *(const v16h*)(Wt + (cc * 16 + r) * DD + k0 + sel * 16);
        #pragma unroll
        for (int cc = 0; cc < 8; ++cc)
            acc[cc] = __builtin_amdgcn_wmma_f32_16x16x32_f16(
                          false, afrag, false, bfr[cc],
                          (short)0, acc[cc], false, false);
    }

    // ---- store C: VGPR v -> row rb + v + sel*8, col cc*16 + r
    #pragma unroll
    for (int cc = 0; cc < 8; ++cc) {
        const int col = cc * 16 + r;
        const float badd = biasOut ? biasOut[col] : 0.0f;
        #pragma unroll
        for (int v = 0; v < 8; ++v) {
            int row = rb + v + sel * 8;
            hOut[(size_t)row * DD + col] = acc[cc][v] + badd;
        }
    }
}

// ---------------- attention dots: a_src/a_dst[b,n] = <h[b,n,:], att> -------
// One wave per (b,n); lane owns 4 contiguous d's (coalesced b128 loads).
__global__ void att_dots(const float* __restrict__ h,
                         const float* __restrict__ att_s,
                         const float* __restrict__ att_d,
                         float* __restrict__ a_src, float* __restrict__ a_dst)
{
    int wid  = (blockIdx.x * blockDim.x + threadIdx.x) >> 5;
    int lane = threadIdx.x & 31;
    if (wid >= MROWS) return;
    const int d = lane * 4;
    float4 hv = *(const float4*)(h + (size_t)wid * DD + d);
    float4 as = *(const float4*)(att_s + d);
    float4 ad = *(const float4*)(att_d + d);
    float s1 = hv.x * as.x + hv.y * as.y + hv.z * as.z + hv.w * as.w;
    float s2 = hv.x * ad.x + hv.y * ad.y + hv.z * ad.z + hv.w * ad.w;
    #pragma unroll
    for (int off = 16; off > 0; off >>= 1) {
        s1 += __shfl_down(s1, off, 32);
        s2 += __shfl_down(s2, off, 32);
    }
    if (lane == 0) { a_src[wid] = s1; a_dst[wid] = s2; }
}

__device__ __forceinline__ float edge_logit(const float* a_src, const float* a_dst,
                                            int b, int sN, int dN) {
    float ae = a_src[b * NN + sN] + a_dst[b * NN + dN];
    return ae > 0.0f ? ae : 0.01f * ae;   // leaky_relu
}

// ---------------- segment max (ordered-uint atomic max) --------------------
__global__ void edge_max(const int* __restrict__ src, const int* __restrict__ dst,
                         const float* __restrict__ a_src, const float* __restrict__ a_dst,
                         unsigned* __restrict__ mmax, int Etot)
{
    int i = blockIdx.x * blockDim.x + threadIdx.x;
    if (i >= Etot * BB) return;
    int e = i >> 2, b = i & 3;
    int sN = src[e], dN = dst[e];
    float ae = edge_logit(a_src, a_dst, b, sN, dN);
    atomicMax(&mmax[dN * BB + b], f2mono(ae));
}

// ---------------- segment sum of exp(ae - m) -------------------------------
__global__ void edge_sum(const int* __restrict__ src, const int* __restrict__ dst,
                         const float* __restrict__ a_src, const float* __restrict__ a_dst,
                         const unsigned* __restrict__ mmax, float* __restrict__ ssum,
                         int Etot)
{
    int i = blockIdx.x * blockDim.x + threadIdx.x;
    if (i >= Etot * BB) return;
    int e = i >> 2, b = i & 3;
    int sN = src[e], dN = dst[e];
    float ae = edge_logit(a_src, a_dst, b, sN, dN);
    float ev = __expf(ae - mono2f(mmax[dN * BB + b]));
    atomicAdd(&ssum[dN * BB + b], ev);
}

// ---------------- message scatter: out[b,dst,:] += alpha * h[b,src,:] ------
// One wave per (edge,b); lane owns 4 contiguous d's -> coalesced b128 gather
// of h[src] + 4 native global_atomic_add_f32 into out[dst].
__global__ void edge_scatter(const int* __restrict__ src, const int* __restrict__ dst,
                             const float* __restrict__ a_src, const float* __restrict__ a_dst,
                             const unsigned* __restrict__ mmax, const float* __restrict__ ssum,
                             const float* __restrict__ h, float* __restrict__ out, int Etot)
{
    int wid  = (blockIdx.x * blockDim.x + threadIdx.x) >> 5;
    int lane = threadIdx.x & 31;
    if (wid >= Etot * BB) return;
    int e = wid >> 2, b = wid & 3;
    int sN = src[e], dN = dst[e];
    float ae    = edge_logit(a_src, a_dst, b, sN, dN);
    float alpha = __expf(ae - mono2f(mmax[dN * BB + b])) / ssum[dN * BB + b];
    const int d = lane * 4;
    const float* hp = h   + ((size_t)b * NN + sN) * DD + d;
    float*       op = out + ((size_t)b * NN + dN) * DD + d;
    float4 hv = *(const float4*)hp;
    atomicAdd(op + 0, alpha * hv.x);
    atomicAdd(op + 1, alpha * hv.y);
    atomicAdd(op + 2, alpha * hv.z);
    atomicAdd(op + 3, alpha * hv.w);
}

// ---------------- per-layer epilogue: x = relu(out + bias) (H=1 mean) ------
__global__ void bias_relu_v4(float4* __restrict__ out, const float4* __restrict__ bias,
                             int n4)
{
    int i = blockIdx.x * blockDim.x + threadIdx.x;
    if (i >= n4) return;
    float4 v = out[i];
    float4 bv = bias[i & (DD / 4 - 1)];
    v.x = fmaxf(v.x + bv.x, 0.f);
    v.y = fmaxf(v.y + bv.y, 0.f);
    v.z = fmaxf(v.z + bv.z, 0.f);
    v.w = fmaxf(v.w + bv.w, 0.f);
    out[i] = v;
}

// ---------------- head: state_emb[b,d] = sum_n x[b,n,d] --------------------
__global__ void node_sum(const float* __restrict__ x, float* __restrict__ semb)
{
    int i = blockIdx.x * blockDim.x + threadIdx.x;
    if (i >= BB * DD) return;
    int b = i >> 7, d = i & (DD - 1);
    const float* p = x + (size_t)b * NN * DD + d;
    float s = 0.f;
    for (int n = 0; n < NN; ++n) s += p[(size_t)n * DD];
    semb[i] = s;
}

// ---------------- head: beta_state = state_emb @ W1 + b1 -------------------
__global__ void small_gemm(const float* __restrict__ se, const float* __restrict__ W1,
                           const float* __restrict__ b1, float* __restrict__ bs)
{
    int i = blockIdx.x * blockDim.x + threadIdx.x;
    if (i >= BB * DD) return;
    int b = i >> 7, j = i & (DD - 1);
    float s = b1[j];
    #pragma unroll 8
    for (int k = 0; k < DD; ++k) s += se[b * DD + k] * W1[k * DD + j];
    bs[i] = s;
}

// ---------------- head: out = relu(concat) @ W3 + b3 -----------------------
// One wave per (b,n); lane owns 4 contiguous d's.
__global__ void final_head(const float* __restrict__ bs, const float* __restrict__ ba,
                           const float* __restrict__ W3, const float* __restrict__ b3,
                           float* __restrict__ outp)
{
    int wid  = (blockIdx.x * blockDim.x + threadIdx.x) >> 5;
    int lane = threadIdx.x & 31;
    if (wid >= MROWS) return;
    int b = wid / NN;
    const int d = lane * 4;
    float4 v1 = *(const float4*)(bs + b * DD + d);
    float4 v2 = *(const float4*)(ba + (size_t)wid * DD + d);
    float4 wa = *(const float4*)(W3 + d);
    float4 wb = *(const float4*)(W3 + DD + d);
    float s = fmaxf(v1.x, 0.f) * wa.x + fmaxf(v1.y, 0.f) * wa.y
            + fmaxf(v1.z, 0.f) * wa.z + fmaxf(v1.w, 0.f) * wa.w
            + fmaxf(v2.x, 0.f) * wb.x + fmaxf(v2.y, 0.f) * wb.y
            + fmaxf(v2.z, 0.f) * wb.z + fmaxf(v2.w, 0.f) * wb.w;
    #pragma unroll
    for (int off = 16; off > 0; off >>= 1) s += __shfl_down(s, off, 32);
    if (lane == 0) outp[wid] = s + b3[0];
}

// ---------------- host driver (graph-capture safe: launches only) ----------
extern "C" void kernel_launch(void* const* d_in, const int* in_sizes, int n_in,
                              void* d_out, int out_size, void* d_ws, size_t ws_size,
                              hipStream_t stream)
{
    const float* state   = (const float*)d_in[0];   // [B,N]
    const float* struc   = (const float*)d_in[1];   // [N,D]
    const int*   eidx    = (const int*)  d_in[2];   // [2,Etot]
    const float* gat_W   = (const float*)d_in[3];   // [3,D,D]
    const float* att_s   = (const float*)d_in[4];   // [3,1,D]
    const float* att_d   = (const float*)d_in[5];   // [3,1,D]
    const float* gWns    = (const float*)d_in[6];   // [3,1,D]
    const float* gbns    = (const float*)d_in[7];   // [3,D]
    const float* gbias   = (const float*)d_in[8];   // [3,D]
    const float* W1      = (const float*)d_in[9];
    const float* b1      = (const float*)d_in[10];
    const float* W2      = (const float*)d_in[11];
    const float* b2      = (const float*)d_in[12];
    const float* W3      = (const float*)d_in[13];
    const float* b3      = (const float*)d_in[14];
    (void)n_in; (void)out_size; (void)ws_size;

    const int Etot = in_sizes[2] / 2;               // E + N self loops
    const int* src = eidx;
    const int* dst = eidx + Etot;

    // workspace layout (floats); needs ~62.1 MB
    float* ws    = (float*)d_ws;
    float* bufX  = ws;                      // [B,N,D]
    float* bufH  = ws + (size_t)MROWS * DD; // [B,N,D]
    float* bufO  = bufH + (size_t)MROWS * DD;
    float* a_src = bufO + (size_t)MROWS * DD;       // [B*N]
    float* a_dst = a_src + MROWS;                   // [B*N]
    float* ssum  = a_dst + MROWS;                   // [N*B]
    unsigned* mmax = (unsigned*)(ssum + MROWS);     // [N*B]
    float* semb   = ssum + 2 * MROWS;               // [B,D]
    float* bstate = semb + BB * DD;                 // [B,D]

    const int BIG   = MROWS * DD;                   // 5,120,000
    const int gBig4 = (BIG / 4) / 256;              // 5000 (float4 clears/relu)
    const int gEB   = (Etot * BB + 255) / 256;      // edge-thread kernels
    const int gEW   = (Etot * BB + 7) / 8;          // 8 waves / 256-thread block
    const int gNW   = (MROWS + 7) / 8;              // one wave per (b,n)
    const int gSm   = (MROWS + 255) / 256;

    const float* xin = struc;
    int bc = 1;
    float* outs[3] = { bufX, bufO, bufX };

    for (int l = 0; l < 3; ++l) {
        float* h  = bufH;
        float* ob = outs[l];
        fused_gemm<<<MROWS / 64, 128, 0, stream>>>(xin, bc, state,
            gWns + l * DD, gbns + l * DD, gat_W + l * DD * DD, nullptr, h);
        att_dots<<<gNW, 256, 0, stream>>>(h, att_s + l * DD, att_d + l * DD, a_src, a_dst);
        clear_u32<<<gSm, 256, 0, stream>>>(mmax, MROWS);
        clear_f32<<<gSm, 256, 0, stream>>>(ssum, MROWS);
        clear_f32_v4<<<gBig4, 256, 0, stream>>>((float4*)ob, BIG / 4);
        edge_max<<<gEB, 256, 0, stream>>>(src, dst, a_src, a_dst, mmax, Etot);
        edge_sum<<<gEB, 256, 0, stream>>>(src, dst, a_src, a_dst, mmax, ssum, Etot);
        edge_scatter<<<gEW, 256, 0, stream>>>(src, dst, a_src, a_dst, mmax, ssum, h, ob, Etot);
        bias_relu_v4<<<gBig4, 256, 0, stream>>>((float4*)ob,
            (const float4*)(gbias + l * DD), BIG / 4);
        xin = ob; bc = 0;
    }

    // head: x == bufX (layer-2 output)
    node_sum<<<(BB * DD + 255) / 256, 256, 0, stream>>>(xin, semb);
    small_gemm<<<(BB * DD + 255) / 256, 256, 0, stream>>>(semb, W1, b1, bstate);
    // beta_action = x @ W2 + b2 (reuse WMMA GEMM, no state fusion)
    fused_gemm<<<MROWS / 64, 128, 0, stream>>>(xin, 0, nullptr, nullptr, nullptr,
                                               W2, b2, bufH);
    // one wave per (b,n) output element — full MROWS coverage
    final_head<<<gNW, 256, 0, stream>>>(bstate, bufH, W3, b3, (float*)d_out);
}
